// MultiHeadAttention_61194694034164
// MI455X (gfx1250) — compile-verified, matrix operand order checked by reading, not run on hardware
//
#include <hip/hip_runtime.h>
#include <hip/hip_bf16.h>
#include <math.h>

// ---------------------------------------------------------------------------
// MHA with relative-position bias for MI455X (gfx1250, wave32, WMMA).
// All GEMMs: v_wmma_f32_16x16x32_bf16 (bf16 inputs, f32 accumulate).
// ---------------------------------------------------------------------------

typedef __bf16 bf16_t;
typedef __attribute__((ext_vector_type(8)))  __bf16 v8bf;
typedef __attribute__((ext_vector_type(16))) __bf16 v16bf;
typedef __attribute__((ext_vector_type(8)))  float  v8f;

#define T_DIM 1024
#define S_DIM 1024
#define BATCH 8
#define HID   1024
#define NH    16
#define HD    64
#define BH    128          // BATCH*NH
#define RELK  16
#define NRJ   33           // 2*RELK+1

__device__ __forceinline__ v16bf comb(v8bf lo, v8bf hi) {
  return __builtin_shufflevector(lo, hi, 0,1,2,3,4,5,6,7,8,9,10,11,12,13,14,15);
}

// 16x32 bf16 fragment, A-operand lane layout (ISA 7.12.2):
// lane L (0-15): row M=L, K in [h*8, h*8+8) U [16+h*8, 16+h*8+8), h = L>>4.
// Both runs are contiguous 16-byte chunks -> 2x b128 loads per lane.
// B operands are fed from pre-transposed matrices so the same loader applies.
__device__ __forceinline__ v16bf load_frag_bf16(const bf16_t* p, int ld) {
  int lane = threadIdx.x & 31;
  const bf16_t* rp = p + (size_t)(lane & 15) * ld + ((lane >> 4) << 3);
  v8bf lo = *(const v8bf*)rp;
  v8bf hi = *(const v8bf*)(rp + 16);
  return comb(lo, hi);
}

// Same fragment but sourced from f32 memory (on-the-fly bf16 downconvert).
__device__ __forceinline__ v16bf load_frag_f32(const float* p, int ld) {
  int lane = threadIdx.x & 31;
  const float* rp = p + (size_t)(lane & 15) * ld + ((lane >> 4) << 3);
  v16bf o;
#pragma unroll
  for (int i = 0; i < 8; ++i) o[i] = (bf16_t)rp[i];
#pragma unroll
  for (int i = 0; i < 8; ++i) o[8 + i] = (bf16_t)rp[16 + i];
  return o;
}

// ---------------------------------------------------------------------------
// Weight transpose + bf16 convert: WT[n*HID + k] = (bf16)W[k*HID + n]
// ---------------------------------------------------------------------------
__global__ __launch_bounds__(256) void k_wt(const float* __restrict__ W,
                                            bf16_t* __restrict__ WT) {
  int gid = blockIdx.x * 256 + threadIdx.x;      // HID*HID threads exactly
  int n = gid >> 10;
  int k = gid & 1023;
  WT[(size_t)n * HID + k] = (bf16_t)W[(size_t)k * HID + n];
}

// ---------------------------------------------------------------------------
// GEMM Y = X @ W + b on (M=8192, N=1024, K=1024) via bf16 WMMA.
// Block = 4 waves = 128x64 output tile; wave = 32(M) x 64(N):
// per K-step 2 A fragments share 4 B fragments -> 8 WMMAs, halving B traffic
// per FLOP vs 16-row blocking.
// MODE 0: Q proj -> qh  (BH,T,D) bf16, scaled by D^-0.5
// MODE 1: K proj -> kh  (BH,S,D) bf16
// MODE 2: V proj -> vt  (BH,D,S) bf16  (transposed for PV B-operand)
// MODE 3: out proj (A is bf16) -> f32 row-major (T*B, HID)
// ---------------------------------------------------------------------------
template <int MODE>
__global__ __launch_bounds__(128) void k_gemm(const void* __restrict__ Ain,
                                              const bf16_t* __restrict__ BT,
                                              const float* __restrict__ bias,
                                              void* __restrict__ out) {
  const int K = HID;
  int wave = threadIdx.x >> 5;
  int lane = threadIdx.x & 31;
  int row0 = blockIdx.y * 128 + wave * 32;
  int col0 = blockIdx.x * 64;

  v8f acc[2][4] = {};
  for (int kk = 0; kk < K; kk += 32) {
    v16bf a0, a1;
    if (MODE < 3) {
      a0 = load_frag_f32((const float*)Ain + (size_t)row0 * K + kk, K);
      a1 = load_frag_f32((const float*)Ain + (size_t)(row0 + 16) * K + kk, K);
    } else {
      a0 = load_frag_bf16((const bf16_t*)Ain + (size_t)row0 * K + kk, K);
      a1 = load_frag_bf16((const bf16_t*)Ain + (size_t)(row0 + 16) * K + kk, K);
    }
    if (kk + 32 < K) {   // gfx1250 global_prefetch for the next A slab
      __builtin_prefetch((const char*)Ain + ((size_t)row0 * K + kk + 32) * (MODE < 3 ? 4 : 2), 0, 1);
    }
#pragma unroll
    for (int c = 0; c < 4; ++c) {
      v16bf b = load_frag_bf16(BT + (size_t)(col0 + 16 * c) * K + kk, K);
      acc[0][c] = __builtin_amdgcn_wmma_f32_16x16x32_bf16(false, a0, false, b,
                                                          (short)0, acc[0][c], false, false);
      acc[1][c] = __builtin_amdgcn_wmma_f32_16x16x32_bf16(false, a1, false, b,
                                                          (short)0, acc[1][c], false, false);
    }
  }

  // C/D layout: lane half h=lane>>4, cl=lane&15 -> col N=cl, rows M=v+8h.
  int cl = lane & 15, hf = lane >> 4;
#pragma unroll
  for (int r = 0; r < 2; ++r) {
#pragma unroll
    for (int c = 0; c < 4; ++c) {
      int col = col0 + 16 * c + cl;
      float bv = bias[col];
      int hh = col >> 6, d = col & 63;
#pragma unroll
      for (int v = 0; v < 8; ++v) {
        int m = row0 + 16 * r + v + 8 * hf;       // row in (T*B) / (S*B)
        float val = acc[r][c][v] + bv;
        if (MODE == 0) {
          int t = m >> 3, bt = m & 7, n = bt * NH + hh;
          ((bf16_t*)out)[((size_t)n * T_DIM + t) * HD + d] = (bf16_t)(val * 0.125f);
        } else if (MODE == 1) {
          int s = m >> 3, bt = m & 7, n = bt * NH + hh;
          ((bf16_t*)out)[((size_t)n * S_DIM + s) * HD + d] = (bf16_t)val;
        } else if (MODE == 2) {
          int s = m >> 3, bt = m & 7, n = bt * NH + hh;
          ((bf16_t*)out)[((size_t)n * HD + d) * S_DIM + s] = (bf16_t)val;
        } else {
          ((float*)out)[(size_t)m * HID + col] = val;
        }
      }
    }
  }
}

// ---------------------------------------------------------------------------
// rel_bias[n,t,j] = qh[n,t,:] . rel_k_table[j,:]   (qh already scaled)
// ---------------------------------------------------------------------------
__global__ __launch_bounds__(256) void k_relbias(const bf16_t* __restrict__ qh,
                                                 const float* __restrict__ relk,
                                                 float* __restrict__ rb) {
  int gid = blockIdx.x * 256 + threadIdx.x;
  if (gid >= BH * T_DIM * NRJ) return;
  int j = gid % NRJ;
  int nt = gid / NRJ;
  const bf16_t* q = qh + (size_t)nt * HD;
  const float* r = relk + j * HD;
  float s = 0.f;
#pragma unroll
  for (int d = 0; d < HD; ++d) s += (float)q[d] * r[d];
  rb[gid] = s;
}

// ---------------------------------------------------------------------------
// Attention: one wave handles (head n, 16-row T tile).
// Scores staged bf16 in LDS; softmax with cross-lane max/sum; 33 rel-pos
// bucket sums via ds_add_f32; PV through WMMA with P fragments from LDS.
// ---------------------------------------------------------------------------
#define LDP (S_DIM + 8)   // LDS row pad: breaks the 2048B-stride bank pattern

__global__ __launch_bounds__(32) void k_attn(
    const bf16_t* __restrict__ qh, const bf16_t* __restrict__ kh,
    const bf16_t* __restrict__ vt, const float* __restrict__ rb,
    const float* __restrict__ mask, const float* __restrict__ relv,
    bf16_t* __restrict__ oflat) {
  __shared__ bf16_t sP[16][LDP];   // scores, then exp(P)
  __shared__ float  sB[16][NRJ];   // rel-pos bucket sums
  __shared__ float  sL[16];        // row softmax denominators

  int n = blockIdx.y, ttile = blockIdx.x;
  int t0 = ttile * 16;
  int bbat = n >> 4, hh = n & 15;
  int lane = threadIdx.x, hf = lane >> 4, cl = lane & 15;
  int nst = ttile + 1;             // causal: S tiles 0..ttile

  for (int i = lane; i < 16 * NRJ; i += 32) sB[i / NRJ][i % NRJ] = 0.f;

  const bf16_t* qbase = qh + ((size_t)n * T_DIM + t0) * HD;
  v16bf q0 = load_frag_bf16(qbase + 0,  HD);
  v16bf q1 = load_frag_bf16(qbase + 32, HD);

  float rmax[8];
#pragma unroll
  for (int v = 0; v < 8; ++v) rmax[v] = -3.0e38f;

  // ---- Phase 1: scores = Q Kt. Two INDEPENDENT K=32 WMMAs (no D->C chain,
  // avoids the 4-NOP bf16 WMMA->WMMA hazard); halves summed by VALU, which
  // fuses into the rel_bias + mask v_pk_add stream.
  for (int st = 0; st < nst; ++st) {
    int s0 = st * 16;
    const bf16_t* kbase = kh + ((size_t)n * S_DIM + s0) * HD;
    v16bf k0 = load_frag_bf16(kbase + 0,  HD);
    v16bf k1 = load_frag_bf16(kbase + 32, HD);
    v8f c0 = {}, c1 = {};
    c0 = __builtin_amdgcn_wmma_f32_16x16x32_bf16(false, q0, false, k0, (short)0, c0, false, false);
    c1 = __builtin_amdgcn_wmma_f32_16x16x32_bf16(false, q1, false, k1, (short)0, c1, false, false);
    int s = s0 + cl;
#pragma unroll
    for (int v = 0; v < 8; ++v) {
      int row = v + 8 * hf;
      int t = t0 + row;
      int rel = s - t; rel = rel < -RELK ? -RELK : (rel > RELK ? RELK : rel);
      float x = (c0[v] + c1[v])
              + rb[((size_t)n * T_DIM + t) * NRJ + (rel + RELK)]
              + mask[((size_t)bbat * T_DIM + t) * S_DIM + s];
      rmax[v] = fmaxf(rmax[v], x);
      sP[row][s] = (bf16_t)x;
    }
  }
  __syncthreads();

  // ---- row max across the 16 lanes of each half ----
#pragma unroll
  for (int v = 0; v < 8; ++v) {
    rmax[v] = fmaxf(rmax[v], __shfl_xor(rmax[v], 1, 32));
    rmax[v] = fmaxf(rmax[v], __shfl_xor(rmax[v], 2, 32));
    rmax[v] = fmaxf(rmax[v], __shfl_xor(rmax[v], 4, 32));
    rmax[v] = fmaxf(rmax[v], __shfl_xor(rmax[v], 8, 32));
  }

  // ---- Phase 2: exp, row sums, rel-pos bucket sums ----
  float rsum[8];
#pragma unroll
  for (int v = 0; v < 8; ++v) rsum[v] = 0.f;
  for (int st = 0; st < nst; ++st) {
    int s = st * 16 + cl;
#pragma unroll
    for (int v = 0; v < 8; ++v) {
      int row = v + 8 * hf;
      int t = t0 + row;
      float e = __expf((float)sP[row][s] - rmax[v]);
      sP[row][s] = (bf16_t)e;
      rsum[v] += e;
      int rel = s - t; rel = rel < -RELK ? -RELK : (rel > RELK ? RELK : rel);
      atomicAdd(&sB[row][rel + RELK], e);          // ds_add_f32
    }
  }
#pragma unroll
  for (int v = 0; v < 8; ++v) {
    rsum[v] += __shfl_xor(rsum[v], 1, 32);
    rsum[v] += __shfl_xor(rsum[v], 2, 32);
    rsum[v] += __shfl_xor(rsum[v], 4, 32);
    rsum[v] += __shfl_xor(rsum[v], 8, 32);
  }
  if (cl == 0) {
#pragma unroll
    for (int v = 0; v < 8; ++v) sL[v + 8 * hf] = rsum[v];
  }
  if (nst & 1) {                                   // zero the K=32 pad tile
    int s = nst * 16 + cl;
#pragma unroll
    for (int v = 0; v < 8; ++v) sP[v + 8 * hf][s] = (bf16_t)0.f;
  }
  __syncthreads();

  // ---- Phase 3: O = P @ V, P fragments straight from LDS (ds_load_b128) ----
  v8f o[4] = {};
  int nch = (nst + 1) >> 1;
  for (int ch = 0; ch < nch; ++ch) {
    const bf16_t* pp = &sP[cl][ch * 32 + hf * 8];
    v8bf lo = *(const v8bf*)pp;
    v8bf hi = *(const v8bf*)(pp + 16);
    v16bf pf = comb(lo, hi);
#pragma unroll
    for (int c = 0; c < 4; ++c) {
      v16bf vf = load_frag_bf16(vt + ((size_t)n * HD + c * 16) * S_DIM + ch * 32, S_DIM);
      o[c] = __builtin_amdgcn_wmma_f32_16x16x32_bf16(false, pf, false, vf,
                                                     (short)0, o[c], false, false);
    }
  }

  // ---- Phase 4: + bucket @ rel_v_table, normalize, scatter to (T*B, HID) ----
#pragma unroll
  for (int c = 0; c < 4; ++c) {
    int d = c * 16 + cl;
#pragma unroll
    for (int v = 0; v < 8; ++v) {
      int row = v + 8 * hf;
      int t = t0 + row;
      float a = o[c][v];
      for (int j = 0; j < NRJ; ++j) a += sB[row][j] * relv[j * HD + d];
      a /= sL[row];
      oflat[((size_t)(t * BATCH + bbat)) * HID + hh * HD + d] = (bf16_t)a;
    }
  }
}

// ---------------------------------------------------------------------------
extern "C" void kernel_launch(void* const* d_in, const int* in_sizes, int n_in,
                              void* d_out, int out_size, void* d_ws, size_t ws_size,
                              hipStream_t stream) {
  const float* q    = (const float*)d_in[0];
  const float* k    = (const float*)d_in[1];
  const float* v    = (const float*)d_in[2];
  const float* mask = (const float*)d_in[3];
  const float* Wq   = (const float*)d_in[4];
  const float* bq   = (const float*)d_in[5];
  const float* Wk   = (const float*)d_in[6];
  const float* bk   = (const float*)d_in[7];
  const float* Wv   = (const float*)d_in[8];
  const float* bv   = (const float*)d_in[9];
  const float* Wo   = (const float*)d_in[10];
  const float* bo   = (const float*)d_in[11];
  const float* relk = (const float*)d_in[12];
  const float* relv = (const float*)d_in[13];

  char* ws = (char*)d_ws;
  size_t off = 0;
  bf16_t* WqT = (bf16_t*)(ws + off); off += (size_t)HID * HID * 2;
  bf16_t* WkT = (bf16_t*)(ws + off); off += (size_t)HID * HID * 2;
  bf16_t* WvT = (bf16_t*)(ws + off); off += (size_t)HID * HID * 2;
  bf16_t* WoT = (bf16_t*)(ws + off); off += (size_t)HID * HID * 2;
  bf16_t* qh  = (bf16_t*)(ws + off); off += (size_t)BH * T_DIM * HD * 2;
  bf16_t* kh  = (bf16_t*)(ws + off); off += (size_t)BH * S_DIM * HD * 2;
  bf16_t* vt  = (bf16_t*)(ws + off); off += (size_t)BH * HD * S_DIM * 2;
  float*  rb  = (float*)(ws + off);  off += (size_t)BH * T_DIM * NRJ * 4;
  bf16_t* ofl = (bf16_t*)(ws + off); off += (size_t)T_DIM * BATCH * HID * 2;
  (void)ws_size; (void)in_sizes; (void)n_in; (void)out_size;

  k_wt<<<(HID * HID) / 256, 256, 0, stream>>>(Wq, WqT);
  k_wt<<<(HID * HID) / 256, 256, 0, stream>>>(Wk, WkT);
  k_wt<<<(HID * HID) / 256, 256, 0, stream>>>(Wv, WvT);
  k_wt<<<(HID * HID) / 256, 256, 0, stream>>>(Wo, WoT);

  dim3 g(HID / 64, (T_DIM * BATCH) / 128);
  k_gemm<0><<<g, 128, 0, stream>>>(q, WqT, bq, qh);
  k_gemm<1><<<g, 128, 0, stream>>>(k, WkT, bk, kh);
  k_gemm<2><<<g, 128, 0, stream>>>(v, WvT, bv, vt);

  k_relbias<<<(BH * T_DIM * NRJ + 255) / 256, 256, 0, stream>>>(qh, relk, rb);

  k_attn<<<dim3(T_DIM / 16, BH), 32, 0, stream>>>(qh, kh, vt, rb, mask, relv, ofl);

  k_gemm<3><<<g, 128, 0, stream>>>(ofl, WoT, bo, d_out);
}
